// SpatioTemporalGraphAttention_42769284333615
// MI455X (gfx1250) — compile-verified, matrix operand order checked by reading, not run on hardware
//
#include <hip/hip_runtime.h>
#include <hip/hip_bf16.h>

#define NN 30000
#define EE 480000
#define EPLUS (EE + NN)
#define IN_DIM 32
#define HEADS 8
#define CH 64
#define HCW 512
#define LAYERS 3
#define NEG_SLOPE 0.2f

typedef __attribute__((ext_vector_type(16))) __bf16 v16bf;
typedef __attribute__((ext_vector_type(8)))  __bf16 v8bf;
typedef __attribute__((ext_vector_type(8)))  float  v8f;
typedef __attribute__((ext_vector_type(4)))  float  f4;

__device__ __forceinline__ unsigned short f2bf_bits(float f) {
  unsigned int u = __float_as_uint(f);
  u += 0x7FFFu + ((u >> 16) & 1u);          // round-to-nearest-even
  return (unsigned short)(u >> 16);
}

// ---------------- input projection: h = relu(x @ W_in + b_in), also emit bf16 ----
__global__ void k_input_proj(const float* __restrict__ x, const float* __restrict__ W,
                             const float* __restrict__ b, float* __restrict__ h,
                             unsigned short* __restrict__ hb) {
  int idx = blockIdx.x * blockDim.x + threadIdx.x;
  if (idx >= NN * HCW) return;
  int n = idx / HCW, j = idx % HCW;
  float acc = b[j];
  const float* xr = x + n * IN_DIM;
#pragma unroll
  for (int k = 0; k < IN_DIM; ++k) acc = fmaf(xr[k], W[k * HCW + j], acc);
  acc = acc > 0.f ? acc : 0.f;
  h[idx] = acc;
  hb[idx] = f2bf_bits(acc);
}

// ---------------- W_gat -> bf16, transposed (Wt[l][n][k] = W[l][k][n]) ----------
__global__ void k_wt_bf16(const float* __restrict__ Wg, unsigned short* __restrict__ Wt) {
  int idx = blockIdx.x * blockDim.x + threadIdx.x;
  if (idx >= LAYERS * HCW * HCW) return;
  int l = idx / (HCW * HCW);
  int r = idx - l * HCW * HCW;
  int nn = r / HCW, kk = r % HCW;
  Wt[idx] = f2bf_bits(Wg[l * HCW * HCW + kk * HCW + nn]);
}

// ---------------- WMMA GEMM: hp[M,512] = A[M,512](bf16) x W(bf16, pre-transposed) ---
__device__ __forceinline__ v16bf load_frag(const __bf16* p) {
  // p points at (row base + k-block + koff); elements 0..7 -> k+0..7, 8..15 -> k+16..23
  v8bf lo = *(const v8bf*)(p);
  v8bf hi = *(const v8bf*)(p + 16);
  v16bf r;
#pragma unroll
  for (int i = 0; i < 8; ++i) { r[i] = lo[i]; r[i + 8] = hi[i]; }
  return r;
}

__global__ void __launch_bounds__(256) k_gemm_wmma(const __bf16* __restrict__ A,
                                                   const __bf16* __restrict__ Bt,
                                                   float* __restrict__ C) {
  int wave = threadIdx.x >> 5;
  int lane = threadIdx.x & 31;
  int m0 = blockIdx.x << 4;      // 16 rows per block
  int n0 = wave << 6;            // 64 cols per wave (8 waves -> 512)
  int row = lane & 15;
  int koff = (lane & 16) ? 8 : 0;
  const __bf16* pa = A + (m0 + row) * HCW + koff;
  const __bf16* pb = Bt + (n0 + row) * HCW + koff;
  v8f c0 = {}, c1 = {}, c2 = {}, c3 = {};
  // 2-stage software pipeline: issue next k-slab's loads before consuming current
  v16bf a  = load_frag(pa);
  v16bf b0 = load_frag(pb);
  v16bf b1 = load_frag(pb + 16 * HCW);
  v16bf b2 = load_frag(pb + 32 * HCW);
  v16bf b3 = load_frag(pb + 48 * HCW);
  for (int k = 32; k < HCW; k += 32) {
    v16bf an  = load_frag(pa + k);
    v16bf bn0 = load_frag(pb + k);
    v16bf bn1 = load_frag(pb + 16 * HCW + k);
    v16bf bn2 = load_frag(pb + 32 * HCW + k);
    v16bf bn3 = load_frag(pb + 48 * HCW + k);
    c0 = __builtin_amdgcn_wmma_f32_16x16x32_bf16(false, a, false, b0, (short)0, c0, false, false);
    c1 = __builtin_amdgcn_wmma_f32_16x16x32_bf16(false, a, false, b1, (short)0, c1, false, false);
    c2 = __builtin_amdgcn_wmma_f32_16x16x32_bf16(false, a, false, b2, (short)0, c2, false, false);
    c3 = __builtin_amdgcn_wmma_f32_16x16x32_bf16(false, a, false, b3, (short)0, c3, false, false);
    a = an; b0 = bn0; b1 = bn1; b2 = bn2; b3 = bn3;
  }
  c0 = __builtin_amdgcn_wmma_f32_16x16x32_bf16(false, a, false, b0, (short)0, c0, false, false);
  c1 = __builtin_amdgcn_wmma_f32_16x16x32_bf16(false, a, false, b1, (short)0, c1, false, false);
  c2 = __builtin_amdgcn_wmma_f32_16x16x32_bf16(false, a, false, b2, (short)0, c2, false, false);
  c3 = __builtin_amdgcn_wmma_f32_16x16x32_bf16(false, a, false, b3, (short)0, c3, false, false);
  // C/D layout: VGPR r -> M = r (lanes 0-15) / 8+r (lanes 16-31), N = lane&15
  int mr = m0 + ((lane >> 4) << 3);
  int nc = lane & 15;
#pragma unroll
  for (int r = 0; r < 8; ++r) {
    float* crow = C + (mr + r) * HCW + n0 + nc;
    crow[0]  = c0[r];
    crow[16] = c1[r];
    crow[32] = c2[r];
    crow[48] = c3[r];
  }
}

// ---------------- per-node attention logits s,d ---------------------------------
__global__ void k_attn_sd(const float* __restrict__ hp, const float* __restrict__ asrc,
                          const float* __restrict__ adst, float* __restrict__ s,
                          float* __restrict__ d) {
  int idx = blockIdx.x * blockDim.x + threadIdx.x;
  if (idx >= NN * HEADS) return;
  int n = idx / HEADS, h = idx % HEADS;
  const float* rowp = hp + n * HCW + h * CH;
  const float* as = asrc + h * CH;
  const float* ad = adst + h * CH;
  float ss = 0.f, dd = 0.f;
#pragma unroll 8
  for (int c = 0; c < CH; ++c) {
    float v = rowp[c];
    ss = fmaf(v, as[c], ss);
    dd = fmaf(v, ad[c], dd);
  }
  s[idx] = ss;
  d[idx] = dd;
}

// ================= CSR construction (once per launch; graph static) =============
__global__ void k_zero_int(int* __restrict__ p, int n) {
  int idx = blockIdx.x * blockDim.x + threadIdx.x;
  if (idx < n) p[idx] = 0;
}

__global__ void k_deg_count(const int* __restrict__ dst, int* __restrict__ deg) {
  int e = blockIdx.x * blockDim.x + threadIdx.x;
  if (e >= EPLUS) return;
  int di = (e < EE) ? dst[e] : (e - EE);
  atomicAdd(&deg[di], 1);
}

// single block, 256 threads: exclusive scan of deg -> row_start, copy to cursor
__global__ void __launch_bounds__(256) k_scan(const int* __restrict__ deg,
                                              int* __restrict__ row_start,
                                              int* __restrict__ cursor) {
  __shared__ int part[256];
  const int CHUNK = (NN + 255) / 256;   // 118
  int t = threadIdx.x;
  int lo = t * CHUNK;
  int hi = lo + CHUNK; if (hi > NN) hi = NN;
  int s = 0;
  for (int i = lo; i < hi; ++i) s += deg[i];
  part[t] = s;
  __syncthreads();
  if (t == 0) {
    int run = 0;
    for (int i = 0; i < 256; ++i) { int v = part[i]; part[i] = run; run += v; }
  }
  __syncthreads();
  int run = part[t];
  for (int i = lo; i < hi; ++i) {
    row_start[i] = run;
    cursor[i] = run;
    run += deg[i];
  }
  if (t == 0) row_start[NN] = EPLUS;
}

__global__ void k_scatter(const int* __restrict__ dst, int* __restrict__ cursor,
                          int* __restrict__ csr_eid) {
  int e = blockIdx.x * blockDim.x + threadIdx.x;
  if (e >= EPLUS) return;
  int di = (e < EE) ? dst[e] : (e - EE);
  int pos = atomicAdd(&cursor[di], 1);
  csr_eid[pos] = e;
}

// ================= atomic-free edge softmax + aggregation ======================
__device__ __forceinline__ f4 leaky4(f4 v) {
  f4 r;
#pragma unroll
  for (int i = 0; i < 4; ++i) r[i] = v[i] > 0.f ? v[i] : NEG_SLOPE * v[i];
  return r;
}

// one thread per edge: eb[e*8+h] = leaky_relu(s[src][h] + d[dst][h])
__global__ void k_edge_logits(const int* __restrict__ src, const int* __restrict__ dst,
                              const float* __restrict__ s, const float* __restrict__ d,
                              float* __restrict__ eb) {
  int e = blockIdx.x * blockDim.x + threadIdx.x;
  if (e >= EPLUS) return;
  int si = (e < EE) ? src[e] : (e - EE);
  int di = (e < EE) ? dst[e] : (e - EE);
  f4 s0 = *(const f4*)(s + si * HEADS);
  f4 s1 = *(const f4*)(s + si * HEADS + 4);
  f4 d0 = *(const f4*)(d + di * HEADS);
  f4 d1 = *(const f4*)(d + di * HEADS + 4);
  *(f4*)(eb + e * HEADS)     = leaky4(s0 + d0);
  *(f4*)(eb + e * HEADS + 4) = leaky4(s1 + d1);
}

// one thread per dst node: segment softmax over its CSR range, eb := alpha
__global__ void k_node_softmax(const int* __restrict__ row_start,
                               const int* __restrict__ csr_eid,
                               float* __restrict__ eb) {
  int n = blockIdx.x * blockDim.x + threadIdx.x;
  if (n >= NN) return;
  int r0 = row_start[n], r1 = row_start[n + 1];
  float ninf = __uint_as_float(0xFF800000u);
  f4 m0 = {ninf, ninf, ninf, ninf}, m1 = m0;
  for (int r = r0; r < r1; ++r) {
    int e = csr_eid[r];
    f4 v0 = *(const f4*)(eb + e * HEADS);
    f4 v1 = *(const f4*)(eb + e * HEADS + 4);
#pragma unroll
    for (int i = 0; i < 4; ++i) { m0[i] = fmaxf(m0[i], v0[i]); m1[i] = fmaxf(m1[i], v1[i]); }
  }
  f4 sum0 = {0.f, 0.f, 0.f, 0.f}, sum1 = sum0;
  for (int r = r0; r < r1; ++r) {
    int e = csr_eid[r];
    f4 v0 = *(const f4*)(eb + e * HEADS);
    f4 v1 = *(const f4*)(eb + e * HEADS + 4);
#pragma unroll
    for (int i = 0; i < 4; ++i) { sum0[i] += __expf(v0[i] - m0[i]); sum1[i] += __expf(v1[i] - m1[i]); }
  }
  f4 inv0, inv1;
#pragma unroll
  for (int i = 0; i < 4; ++i) {
    inv0[i] = 1.f / (sum0[i] + 1e-16f);
    inv1[i] = 1.f / (sum1[i] + 1e-16f);
  }
  for (int r = r0; r < r1; ++r) {
    int e = csr_eid[r];
    f4 v0 = *(const f4*)(eb + e * HEADS);
    f4 v1 = *(const f4*)(eb + e * HEADS + 4);
#pragma unroll
    for (int i = 0; i < 4; ++i) { v0[i] = __expf(v0[i] - m0[i]) * inv0[i]; v1[i] = __expf(v1[i] - m1[i]) * inv1[i]; }
    *(f4*)(eb + e * HEADS)     = v0;
    *(f4*)(eb + e * HEADS + 4) = v1;
  }
}

// one 256-thread block per dst node: h[dst] = relu(sum alpha*hp[src] + b); emit bf16
__global__ void __launch_bounds__(256) k_aggregate(const int* __restrict__ src,
                                                   const int* __restrict__ row_start,
                                                   const int* __restrict__ csr_eid,
                                                   const float* __restrict__ eb,
                                                   const float* __restrict__ hp,
                                                   const float* __restrict__ b,
                                                   float* __restrict__ h,
                                                   unsigned short* __restrict__ hb) {
  int n = blockIdx.x;
  int j = threadIdx.x;
  int c0 = j, c1 = j + 256;
  int h0 = c0 >> 6, h1 = c1 >> 6;
  int r0 = row_start[n], r1 = row_start[n + 1];
  float acc0 = 0.f, acc1 = 0.f;
  for (int r = r0; r < r1; ++r) {
    int e = csr_eid[r];
    int si = (e < EE) ? src[e] : (e - EE);
    const float* hrow = hp + (size_t)si * HCW;
    float a0 = eb[e * HEADS + h0];
    float a1 = eb[e * HEADS + h1];
    acc0 = fmaf(hrow[c0], a0, acc0);
    acc1 = fmaf(hrow[c1], a1, acc1);
  }
  float v0 = acc0 + b[c0]; v0 = v0 > 0.f ? v0 : 0.f;
  float v1 = acc1 + b[c1]; v1 = v1 > 0.f ? v1 : 0.f;
  size_t base = (size_t)n * HCW;
  h[base + c0] = v0;
  h[base + c1] = v1;
  hb[base + c0] = f2bf_bits(v0);
  hb[base + c1] = f2bf_bits(v1);
}

// ---------------- heads: attention softmax + anomaly score ----------------------
__global__ void k_heads(const float* __restrict__ h, const float* __restrict__ Watt,
                        const float* __restrict__ batt, const float* __restrict__ Wh1,
                        const float* __restrict__ bh1, const float* __restrict__ Wh2,
                        const float* __restrict__ bh2, float* __restrict__ out) {
  int n = blockIdx.x * blockDim.x + threadIdx.x;
  if (n >= NN) return;
  float za[HEADS];
  float z1[CH];
#pragma unroll
  for (int hh = 0; hh < HEADS; ++hh) za[hh] = batt[hh];
#pragma unroll
  for (int c = 0; c < CH; ++c) z1[c] = bh1[c];
  const float* hr = h + (size_t)n * HCW;
  for (int k = 0; k < HCW; ++k) {
    float hv = hr[k];
    const float* wa = Watt + k * HEADS;
#pragma unroll
    for (int hh = 0; hh < HEADS; ++hh) za[hh] = fmaf(hv, wa[hh], za[hh]);
    const float* w1 = Wh1 + k * CH;
#pragma unroll
    for (int c = 0; c < CH; ++c) z1[c] = fmaf(hv, w1[c], z1[c]);
  }
  float mx = za[0];
#pragma unroll
  for (int hh = 1; hh < HEADS; ++hh) mx = fmaxf(mx, za[hh]);
  float sum = 0.f;
#pragma unroll
  for (int hh = 0; hh < HEADS; ++hh) { za[hh] = __expf(za[hh] - mx); sum += za[hh]; }
  float inv = 1.f / sum;
#pragma unroll
  for (int hh = 0; hh < HEADS; ++hh) out[NN + n * HEADS + hh] = za[hh] * inv;
  float sc = bh2[0];
#pragma unroll
  for (int c = 0; c < CH; ++c) {
    float v = z1[c] > 0.f ? z1[c] : 0.f;
    sc = fmaf(v, Wh2[c], sc);
  }
  out[n] = 1.f / (1.f + __expf(-sc));
}

extern "C" void kernel_launch(void* const* d_in, const int* in_sizes, int n_in,
                              void* d_out, int out_size, void* d_ws, size_t ws_size,
                              hipStream_t stream) {
  const float* x       = (const float*)d_in[0];
  const int*   ei      = (const int*)d_in[1];
  const float* W_in    = (const float*)d_in[2];
  const float* b_in    = (const float*)d_in[3];
  const float* W_gat   = (const float*)d_in[4];
  const float* att_src = (const float*)d_in[5];
  const float* att_dst = (const float*)d_in[6];
  const float* b_gat   = (const float*)d_in[7];
  const float* W_att   = (const float*)d_in[8];
  const float* b_att   = (const float*)d_in[9];
  const float* W_h1    = (const float*)d_in[10];
  const float* b_h1    = (const float*)d_in[11];
  const float* W_h2    = (const float*)d_in[12];
  const float* b_h2    = (const float*)d_in[13];
  float* out = (float*)d_out;

  const int* srcp = ei;
  const int* dstp = ei + EE;

  // workspace carve-up (all region offsets stay 16B aligned)
  char* ws = (char*)d_ws;
  float*          h     = (float*)ws;          ws += (size_t)NN * HCW * 4;   // 61.44 MB
  float*          hp    = (float*)ws;          ws += (size_t)NN * HCW * 4;   // 61.44 MB
  unsigned short* hb    = (unsigned short*)ws; ws += (size_t)NN * HCW * 2;   // 30.72 MB
  unsigned short* Wt    = (unsigned short*)ws; ws += (size_t)LAYERS * HCW * HCW * 2;
  float*          sbuf  = (float*)ws;          ws += (size_t)NN * HEADS * 4;
  float*          dbuf  = (float*)ws;          ws += (size_t)NN * HEADS * 4;
  float*          ebuf  = (float*)ws;          ws += (size_t)EPLUS * HEADS * 4;
  int*            degc  = (int*)ws;            ws += (size_t)NN * 4;         // deg, then cursor
  int*            rowst = (int*)ws;            ws += (size_t)(NN + 4) * 4;   // NN+1 used
  int*            csre  = (int*)ws;            ws += (size_t)EPLUS * 4;

  const int T = 256;
  auto cdiv = [](long long a, long long b) { return (unsigned int)((a + b - 1) / b); };

  // ---- one-time: bf16 weights, input projection, CSR build ----
  k_wt_bf16<<<cdiv((long long)LAYERS * HCW * HCW, T), T, 0, stream>>>(W_gat, Wt);
  k_input_proj<<<cdiv((long long)NN * HCW, T), T, 0, stream>>>(x, W_in, b_in, h, hb);
  k_zero_int<<<cdiv(NN, T), T, 0, stream>>>(degc, NN);
  k_deg_count<<<cdiv(EPLUS, T), T, 0, stream>>>(dstp, degc);
  k_scan<<<1, 256, 0, stream>>>(degc, rowst, degc /* becomes cursor */);
  k_scatter<<<cdiv(EPLUS, T), T, 0, stream>>>(dstp, degc, csre);

  for (int l = 0; l < LAYERS; ++l) {
    const __bf16* Wtl = (const __bf16*)(Wt + (size_t)l * HCW * HCW);
    // hp = h @ W_gat[l]   (bf16 WMMA, f32 accumulate, software-pipelined)
    k_gemm_wmma<<<NN / 16, 256, 0, stream>>>((const __bf16*)hb, Wtl, hp);
    // per-node logits
    k_attn_sd<<<cdiv((long long)NN * HEADS, T), T, 0, stream>>>(
        hp, att_src + (size_t)l * HEADS * CH, att_dst + (size_t)l * HEADS * CH, sbuf, dbuf);
    // atomic-free segment softmax + aggregation via CSR
    k_edge_logits<<<cdiv(EPLUS, T), T, 0, stream>>>(srcp, dstp, sbuf, dbuf, ebuf);
    k_node_softmax<<<cdiv(NN, T), T, 0, stream>>>(rowst, csre, ebuf);
    k_aggregate<<<NN, 256, 0, stream>>>(srcp, rowst, csre, ebuf, hp,
                                        b_gat + (size_t)l * HCW, h, hb);
  }

  // heads: softmax attention weights + sigmoid anomaly score
  k_heads<<<cdiv(NN, T), T, 0, stream>>>(h, W_att, b_att, W_h1, b_h1, W_h2, b_h2, out);
}
